// CausalWanSelfAttention_8589934592263
// MI455X (gfx1250) — matrix operand order repaired and checked by reading, not compiled, not run on hardware
//
#include <hip/hip_runtime.h>

// ---------------- CDNA5 WMMA types ----------------
typedef __attribute__((ext_vector_type(16))) __bf16 v16bf;
typedef __attribute__((ext_vector_type(8)))  float  v8f;

// Problem constants (fixed by reference setup_inputs)
#define DIMC   1536
#define S_TOT  32760
#define NH     12
#define HD     128
#define FF     21
#define GHT    30
#define GWD    52
#define FLN    (FF*GWD)     // 1092
#define FLP    1096         // FLN padded to multiple of 8
#define AIN    (FF*GHT)     // 630
#define AIP    632          // AIN padded to multiple of 8
#define HKJ    (NH*GHT*GWD) // 18720
#define SS_SCALE 0.2973017787506803f  // (HD^-0.5)^0.5 = 128^-0.25

__device__ __forceinline__ __bf16 f2bf(float f) {
  unsigned u = __builtin_bit_cast(unsigned, f);
  u += 0x7FFFu + ((u >> 16) & 1u);              // round-to-nearest-even
  unsigned short h = (unsigned short)(u >> 16);
  return __builtin_bit_cast(__bf16, h);
}

// ---------------- Generic batched WMMA GEMM ----------------
// C[b1,b2][m,n] = sum_k A[b1,b2][m,k] * B[b1,b2][k,n] (+bias[n])
// B's k/n and C's m may be composite: idx -> (idx/div)*s1 + (idx%div)*s2.
// AV: A k-dim contiguous (sAk==1), rows 16B-aligned -> b128 loads.
// BV: B k-dim contiguous (sBk2==1, simple), cols 16B-aligned -> b128 loads.
struct GemmParams {
  const __bf16* A; const __bf16* B; float* C; __bf16* Cbf; const float* bias;
  int M, N, Kd, b2, batches;
  int aVec, bVec;
  long sAb1, sAb2, sBb1, sBb2, sCb1, sCb2;
  long sAm, sAk;
  int divK; long sBk1, sBk2;
  int divN; long sBn1, sBn2;
  int divM; long sCm1, sCm2;
  long sCn;
};

__device__ __forceinline__ long coff(int idx, int dv, long s1, long s2) {
  return (long)(idx / dv) * s1 + (long)(idx % dv) * s2;
}

// block = 128 threads = 4 waves; each wave computes one 32(M) x 64(N) tile
// (2 M-subtiles x 4 N-subtiles = 8 WMMA per 32-K step; B fragments reused x2).
template <int AV, int BV>
__global__ void wmma_gemm_kernel(GemmParams p) {
  const int wave   = threadIdx.x >> 5;
  const int lane   = threadIdx.x & 31;
  const int laneHi = lane >> 4;      // wave half selects K sub-range (contiguous 16)
  const int lanelo = lane & 15;      // row (A) / col (B,C) within tile
  const int n0 = (blockIdx.x * 4 + wave) * 64;
  if (n0 >= p.N) return;
  const int m0 = blockIdx.y * 32;
  const int b1  = blockIdx.z / p.b2;
  const int b2i = blockIdx.z % p.b2;
  const long Ab = (long)b1 * p.sAb1 + (long)b2i * p.sAb2;
  const long Bb = (long)b1 * p.sBb1 + (long)b2i * p.sBb2;
  const long Cb = (long)b1 * p.sCb1 + (long)b2i * p.sCb2;

  v8f acc[2][4] = {};

  // Clamp M/N addresses: out-of-range A rows / B cols only feed C entries that
  // are never stored, so no zero-fill is needed for them. Only K needs guarding.
  long aRow[2];
  #pragma unroll
  for (int mh = 0; mh < 2; ++mh) {
    int amr = m0 + mh * 16 + lanelo;
    int am  = (amr < p.M) ? amr : (p.M - 1);
    aRow[mh] = Ab + (long)am * p.sAm;
  }
  const int klo   = laneHi * 16;   // A: lane-half K offset (contiguous 16)
  const int kboff = laneHi * 16;   // B: lane-half K offset (contiguous 16)

  long bnOff[4];
  #pragma unroll
  for (int t = 0; t < 4; ++t) {
    int n = n0 + t * 16 + lanelo;
    if (n >= p.N) n = p.N - 1;
    bnOff[t] = coff(n, p.divN, p.sBn1, p.sBn2);
  }

  const int ktEnd = p.Kd & ~31;

  for (int kt = 0; kt < ktEnd; kt += 32) {
    v16bf afrag[2];
    #pragma unroll
    for (int mh = 0; mh < 2; ++mh) {
      if (AV) {
        union { v16bf v; uint4 q[2]; } ua;
        const __bf16* ap = p.A + aRow[mh] + kt + klo;
        ua.q[0] = *(const uint4*)(ap);        // elements 0..7  : K = kt+klo+0..7
        ua.q[1] = *(const uint4*)(ap + 8);    // elements 8..15 : K = kt+klo+8..15
        afrag[mh] = ua.v;
        if (kt + 32 < ktEnd) __builtin_prefetch(ap + 32, 0, 3);
      } else {
        #pragma unroll
        for (int e = 0; e < 16; ++e) {
          int kk = kt + e + klo;
          afrag[mh][e] = p.A[aRow[mh] + (long)kk * p.sAk];
        }
      }
    }
    #pragma unroll
    for (int t = 0; t < 4; ++t) {
      v16bf bfrag;
      if (BV) {
        union { v16bf v; uint4 q[2]; } ub;
        const __bf16* bp = p.B + Bb + bnOff[t] + kt + kboff;
        ub.q[0] = *(const uint4*)(bp);      // elements 0..7  : K = kt+kboff+0..7
        ub.q[1] = *(const uint4*)(bp + 8);  // elements 8..15 : K = kt+kboff+8..15
        bfrag = ub.v;
      } else {
        #pragma unroll
        for (int e = 0; e < 16; ++e) {
          int kk = kt + e + kboff;
          bfrag[e] = p.B[Bb + coff(kk, p.divK, p.sBk1, p.sBk2) + bnOff[t]];
        }
      }
      #pragma unroll
      for (int mh = 0; mh < 2; ++mh)
        acc[mh][t] = __builtin_amdgcn_wmma_f32_16x16x32_bf16(
            false, afrag[mh], false, bfrag, (short)0, acc[mh][t], false, false);
    }
  }

  // K tail: clamp + select zero (no divergent/predicated loads)
  if (ktEnd < p.Kd) {
    const int kt = ktEnd;
    const __bf16 zero = __builtin_bit_cast(__bf16, (unsigned short)0);
    v16bf afrag[2];
    #pragma unroll
    for (int mh = 0; mh < 2; ++mh)
      #pragma unroll
      for (int e = 0; e < 16; ++e) {
        int kk = kt + e + klo;
        int kkc = (kk < p.Kd) ? kk : (p.Kd - 1);
        __bf16 v = p.A[aRow[mh] + (long)kkc * p.sAk];
        afrag[mh][e] = (kk < p.Kd) ? v : zero;
      }
    #pragma unroll
    for (int t = 0; t < 4; ++t) {
      v16bf bfrag;
      #pragma unroll
      for (int e = 0; e < 16; ++e) {
        int kk = kt + e + kboff;
        int kkc = (kk < p.Kd) ? kk : (p.Kd - 1);
        __bf16 v = p.B[Bb + coff(kkc, p.divK, p.sBk1, p.sBk2) + bnOff[t]];
        bfrag[e] = (kk < p.Kd) ? v : zero;
      }
      #pragma unroll
      for (int mh = 0; mh < 2; ++mh)
        acc[mh][t] = __builtin_amdgcn_wmma_f32_16x16x32_bf16(
            false, afrag[mh], false, bfrag, (short)0, acc[mh][t], false, false);
    }
  }

  // C/D layout: VGPR r -> M = tile_m + r + 8*laneHi, N = n-tile + lanelo
  #pragma unroll
  for (int t = 0; t < 4; ++t) {
    int n = n0 + t * 16 + lanelo;
    if (n >= p.N) continue;
    float bv = p.bias ? p.bias[n] : 0.0f;
    #pragma unroll
    for (int mh = 0; mh < 2; ++mh)
      #pragma unroll
      for (int r = 0; r < 8; ++r) {
        int m = m0 + mh * 16 + r + 8 * laneHi;
        if (m >= p.M) continue;
        long off = Cb + coff(m, p.divM, p.sCm1, p.sCm2) + (long)n * p.sCn;
        float v = acc[mh][t][r] + bv;
        p.C[off] = v;
        if (p.Cbf) p.Cbf[off] = f2bf(v);
      }
  }
}

// ---------------- conversion / packing ----------------
__global__ void convert_kernel(const float* src, __bf16* dst, long n) {
  long i = (long)blockIdx.x * 256 + threadIdx.x;
  if (i < n) dst[i] = f2bf(src[i]);
}

__global__ void fill_kernel(float* p, float v, int n) {
  int i = blockIdx.x * 256 + threadIdx.x;
  if (i < n) p[i] = v;
}

// Kt/Vt layout [h][k][d][FLP]: Kt[...fl] = K[f,k,l,h,d] with fl = f*GWD + l
__global__ void pack_kt_kernel(const float* Kf, __bf16* Kt) {
  long i = (long)blockIdx.x * 256 + threadIdx.x;
  if (i >= (long)NH * GHT * HD * FLP) return;
  int fl = (int)(i % FLP);
  long r = i / FLP;
  int d = (int)(r % HD); r /= HD;
  int k = (int)(r % GHT);
  int h = (int)(r / GHT);
  float v = 0.0f;
  if (fl < FLN) {
    int f = fl / GWD, l = fl % GWD;
    v = Kf[((long)(f * GHT + k) * GWD + l) * DIMC + h * HD + d];
  }
  Kt[i] = f2bf(v);
}

// Qt layout [h][j][d][AIP]: Qt[...ai] = Q[a,i,j,h,d] with ai = a*GHT + i
__global__ void pack_qt_kernel(const float* Qf, __bf16* Qt) {
  long idx = (long)blockIdx.x * 256 + threadIdx.x;
  if (idx >= (long)NH * GWD * HD * AIP) return;
  int ai = (int)(idx % AIP);
  long r = idx / AIP;
  int d = (int)(r % HD); r /= HD;
  int j = (int)(r % GWD);
  int h = (int)(r / GWD);
  float v = 0.0f;
  if (ai < AIN) {
    int a = ai / GHT, ii = ai % GHT;
    v = Qf[((long)(a * GHT + ii) * GWD + j) * DIMC + h * HD + d];
  }
  Qt[idx] = f2bf(v);
}

// ---------------- RMSNorm + 3D RoPE (+ sm_scale^0.5), in-place, plus bf16 copy ----------------
__global__ void norm_rope_kernel(float* X, __bf16* Xbf, const float* g,
                                 const float* fcos, const float* fsin, float scale) {
  int row = blockIdx.x;
  int tid = threadIdx.x;
  __shared__ float sm[256];
  float v[6];
  long base = (long)row * DIMC;
  float acc = 0.0f;
  #pragma unroll
  for (int i = 0; i < 6; ++i) { v[i] = X[base + tid * 6 + i]; acc += v[i] * v[i]; }
  sm[tid] = acc; __syncthreads();
  for (int s = 128; s > 0; s >>= 1) { if (tid < s) sm[tid] += sm[tid + s]; __syncthreads(); }
  float inv = rsqrtf(sm[0] / (float)DIMC + 1e-6f);
  __syncthreads();

  int t_f = row / (GHT * GWD);
  int rem = row % (GHT * GWD);
  int hh = rem / GWD, ww = rem % GWD;

  #pragma unroll
  for (int i = 0; i < 6; ++i) v[i] = v[i] * inv * g[tid * 6 + i];

  #pragma unroll
  for (int pi = 0; pi < 3; ++pi) {        // 3 complete (even,odd) pairs per thread
    int c0 = tid * 6 + 2 * pi;
    int d  = c0 % HD;
    int ph = d >> 1;                      // 0..63 ; splits 22/21/21
    int pos = (ph < 22) ? t_f : (ph < 43 ? hh : ww);
    float fc  = fcos[pos * 64 + ph];
    float fsn = fsin[pos * 64 + ph];
    float xr = v[2 * pi], xi = v[2 * pi + 1];
    v[2 * pi]     = xr * fc - xi * fsn;
    v[2 * pi + 1] = xr * fsn + xi * fc;
  }
  #pragma unroll
  for (int i = 0; i < 6; ++i) {
    float y = v[i] * scale;
    X[base + tid * 6 + i]   = y;
    Xbf[base + tid * 6 + i] = f2bf(y);
  }
}

// aR0[i,j,h,d] = sum_a Q[a,i,j,h,d]
__global__ void sum_frames_kernel(const float* Q, float* aR, __bf16* aRbf) {
  int i = blockIdx.x * 256 + threadIdx.x;
  if (i >= GHT * GWD * NH * HD) return;
  float s = 0.0f;
  #pragma unroll
  for (int a = 0; a < FF; ++a) s += Q[(long)a * (GHT * GWD * NH * HD) + i];
  aR[i] = s; aRbf[i] = f2bf(s);
}

// softmax over (f,l)=1092 per (h,k,j); z = bR*min(1/(cR+eps),1e4); cL = E[z']-log(denom)
// R f32 rows are FLN long; Rbf rows are FLP long (padded for b128 loads).
__global__ void softmax_R_kernel(float* R, __bf16* Rbf, const float* cR, float* cL) {
  int rid = blockIdx.x;                    // h*1560 + k*52 + j
  int h = rid / (GHT * GWD);
  int rem = rid % (GHT * GWD);
  int k = rem / GWD, j = rem % GWD;
  long rowi = (long)(h * GHT + k) * GWD + j;
  long baseF = rowi * FLN;
  long baseB = rowi * FLP;
  float sc = fminf(1.0f / (cR[rid] + 1e-6f), 10000.0f);
  int tid = threadIdx.x;
  __shared__ float sm[256];
  float z[5], ez[5];
  float mx = -3.4e38f;
  #pragma unroll
  for (int i = 0; i < 5; ++i) {
    int c = tid + i * 256;
    if (c < FLN) { z[i] = R[baseF + c] * sc; mx = fmaxf(mx, z[i]); } else z[i] = -3.4e38f;
  }
  sm[tid] = mx; __syncthreads();
  for (int s = 128; s > 0; s >>= 1) { if (tid < s) sm[tid] = fmaxf(sm[tid], sm[tid + s]); __syncthreads(); }
  mx = sm[0]; __syncthreads();
  float se = 0.0f, sez = 0.0f;
  #pragma unroll
  for (int i = 0; i < 5; ++i) {
    int c = tid + i * 256;
    if (c < FLN) { float zs = z[i] - mx; float e = expf(zs); ez[i] = e; se += e; sez += e * zs; }
    else ez[i] = 0.0f;
  }
  sm[tid] = se; __syncthreads();
  for (int s = 128; s > 0; s >>= 1) { if (tid < s) sm[tid] += sm[tid + s]; __syncthreads(); }
  se = sm[0]; __syncthreads();
  sm[tid] = sez; __syncthreads();
  for (int s = 128; s > 0; s >>= 1) { if (tid < s) sm[tid] += sm[tid + s]; __syncthreads(); }
  sez = sm[0]; __syncthreads();
  float dinv = 1.0f / se;
  #pragma unroll
  for (int i = 0; i < 5; ++i) {
    int c = tid + i * 256;
    if (c < FLN) { float r = ez[i] * dinv; R[baseF + c] = r; Rbf[baseB + c] = f2bf(r); }
  }
  if (tid == 0) cL[h * (GWD * GHT) + j * GHT + k] = sez * dinv - logf(se);
}

// softmax over k=30 of (bL - cL) per (h,j,(a,i)); L f32 rows AIN, Lbf rows AIP
__global__ void softmax_L_kernel(float* L, __bf16* Lbf, const float* cL) {
  int bx = blockIdx.x;                     // h*52 + j
  int h = bx / GWD, j = bx % GWD;
  long baseF = (long)(h * GWD + j) * GHT * AIN;
  long baseB = (long)(h * GWD + j) * GHT * AIP;
  int clb = h * (GWD * GHT) + j * GHT;
  for (int n = threadIdx.x; n < AIN; n += 256) {
    float mx = -3.4e38f;
    for (int k = 0; k < GHT; ++k) mx = fmaxf(mx, L[baseF + (long)k * AIN + n] - cL[clb + k]);
    float s = 0.0f;
    for (int k = 0; k < GHT; ++k) s += expf(L[baseF + (long)k * AIN + n] - cL[clb + k] - mx);
    float si = 1.0f / s;
    for (int k = 0; k < GHT; ++k) {
      float e = expf(L[baseF + (long)k * AIN + n] - cL[clb + k] - mx) * si;
      L[baseF + (long)k * AIN + n] = e;
      Lbf[baseB + (long)k * AIP + n] = f2bf(e);
    }
  }
}

// cR[h,k,j] = sum over (a,i) of L[h,j,k,(a,i)]
__global__ void rowsum_kernel(const float* L, float* cR) {
  int rid = blockIdx.x;                    // h*1560 + k*52 + j
  int h = rid / (GHT * GWD);
  int rem = rid % (GHT * GWD);
  int k = rem / GWD, j = rem % GWD;
  long base = (((long)h * GWD + j) * GHT + k) * AIN;
  __shared__ float sm[256];
  float s = 0.0f;
  for (int n = threadIdx.x; n < AIN; n += 256) s += L[base + n];
  sm[threadIdx.x] = s; __syncthreads();
  for (int t = 128; t > 0; t >>= 1) { if (threadIdx.x < t) sm[threadIdx.x] += sm[threadIdx.x + t]; __syncthreads(); }
  if (threadIdx.x == 0) cR[rid] = sm[0];
}

// ---------------- host side ----------------
extern "C" void kernel_launch(void* const* d_in, const int* in_sizes, int n_in,
                              void* d_out, int out_size, void* d_ws, size_t ws_size,
                              hipStream_t stream) {
  const float* x    = (const float*)d_in[0];
  const float* wq   = (const float*)d_in[1];
  const float* bq   = (const float*)d_in[2];
  const float* wk   = (const float*)d_in[3];
  const float* bk   = (const float*)d_in[4];
  const float* wv   = (const float*)d_in[5];
  const float* bv   = (const float*)d_in[6];
  const float* wo   = (const float*)d_in[7];
  const float* bo   = (const float*)d_in[8];
  const float* gq   = (const float*)d_in[9];
  const float* gk   = (const float*)d_in[10];
  const float* fcos = (const float*)d_in[11];
  const float* fsin = (const float*)d_in[12];
  float* out = (float*)d_out;

  char* cur = (char*)d_ws;
  auto alloc = [&](size_t elems, size_t esz) -> void* {
    void* p = (void*)cur;
    cur += (elems * esz + 255) & ~(size_t)255;
    return p;
  };

  const long SD  = (long)S_TOT * DIMC;           // 50,319,360
  const long AR  = (long)GHT * GWD * NH * HD;    // 2,396,160
  const long RNF = (long)NH * GHT * GWD * FLN;   // f32 R
  const long RNP = (long)NH * GHT * GWD * FLP;   // padded bf16 R
  const long LNF = (long)NH * GWD * GHT * AIN;
  const long LNP = (long)NH * GWD * GHT * AIP;
  const long KTN = (long)NH * GHT * HD * FLP;    // Kt / Vt
  const long QTN = (long)NH * GWD * HD * AIP;    // Qt

  __bf16* xbf   = (__bf16*)alloc(SD, 2);
  __bf16* wcat  = (__bf16*)alloc((long)3 * DIMC * DIMC, 2);  // [n][k], n in 0..4607
  __bf16* woBf  = (__bf16*)alloc((long)DIMC * DIMC, 2);      // [n][k]
  float*  Qf    = (float*)alloc(SD, 4);
  float*  Kf    = (float*)alloc(SD, 4);
  float*  Vf    = (float*)alloc(SD, 4);
  __bf16* Qbf   = (__bf16*)alloc(SD, 2);
  __bf16* Kbf   = (__bf16*)alloc(SD, 2);
  __bf16* Vbf   = (__bf16*)alloc(SD, 2);
  __bf16* Kt    = (__bf16*)alloc(KTN, 2);
  __bf16* Vt    = (__bf16*)alloc(KTN, 2);
  __bf16* Qt    = (__bf16*)alloc(QTN, 2);
  float*  aR    = (float*)alloc(AR, 4);
  __bf16* aRbf  = (__bf16*)alloc(AR, 2);
  float*  Rw    = (float*)alloc(RNF, 4);
  __bf16* Rbf   = (__bf16*)alloc(RNP, 2);
  float*  aL    = (float*)alloc(AR, 4);
  __bf16* aLbf  = (__bf16*)alloc(AR, 2);
  float*  bLw   = (float*)alloc(LNF, 4);
  __bf16* Lbf   = (__bf16*)alloc(LNP, 2);
  float*  Yw    = (float*)alloc(AR, 4);
  __bf16* Ybf   = (__bf16*)alloc(AR, 2);
  float*  attn  = (float*)alloc(SD, 4);
  __bf16* attnbf= (__bf16*)alloc(SD, 2);
  float*  cRb   = (float*)alloc(HKJ, 4);
  float*  cLb   = (float*)alloc(HKJ, 4);

  auto mk = [&]() { GemmParams p{}; p.divK = 1 << 30; p.divN = 1 << 30; p.divM = 1 << 30;
                    p.b2 = 1; p.batches = 1; return p; };
  auto run = [&](const GemmParams& p) {
    dim3 g((p.N + 255) / 256, (p.M + 31) / 32, p.batches);
    if (p.aVec && p.bVec) wmma_gemm_kernel<1, 1><<<g, dim3(128), 0, stream>>>(p);
    else                  wmma_gemm_kernel<0, 0><<<g, dim3(128), 0, stream>>>(p);
  };
  auto conv = [&](const float* s, __bf16* d, long n) {
    convert_kernel<<<dim3((unsigned)((n + 255) / 256)), dim3(256), 0, stream>>>(s, d, n);
  };

  // strides in Q/K/V (a,i,j,h,d) layout
  const long sQa = (long)GHT * GWD * NH * HD;  // 2,396,160
  const long sQi = (long)GWD * NH * HD;        // 79,872
  const long sQj = (long)NH * HD;              // 1,536
  const long sQh = HD;                         // 128
  const long sRhF = (long)GHT * GWD * FLN;     // f32 R strides
  const long sRkF = (long)GWD * FLN;
  const long sRhP = (long)GHT * GWD * FLP;     // bf16 R strides
  const long sRkP = (long)GWD * FLP;
  const long sLhF = (long)GWD * GHT * AIN;
  const long sLjF = (long)GHT * AIN;
  const long sLhP = (long)GWD * GHT * AIP;
  const long sLjP = (long)GHT * AIP;

  // 1) packing: x and weights (weights stay [n][k] -> straight converts, K-contiguous B)
  conv(x, xbf, SD);
  conv(wq, wcat, (long)DIMC * DIMC);
  conv(wk, wcat + (long)DIMC * DIMC, (long)DIMC * DIMC);
  conv(wv, wcat + (long)2 * DIMC * DIMC, (long)DIMC * DIMC);
  conv(wo, woBf, (long)DIMC * DIMC);

  // 2) projections: Q/K/V = x @ W.T + b   (A,B both K-contiguous -> b128 loads)
  {
    GemmParams p = mk();
    p.M = S_TOT; p.N = DIMC; p.Kd = DIMC;
    p.A = xbf; p.sAm = DIMC; p.sAk = 1; p.aVec = 1;
    p.sBk2 = 1; p.sBn2 = DIMC; p.bVec = 1;
    p.sCm2 = DIMC; p.sCn = 1;
    p.B = wcat;                           p.C = Qf; p.bias = bq; run(p);
    p.B = wcat + (long)DIMC * DIMC;       p.C = Kf; p.bias = bk; run(p);
    p.B = wcat + (long)2 * DIMC * DIMC;   p.C = Vf; p.bias = bv; p.Cbf = Vbf; run(p);
  }

  // 3) rmsnorm + rope + ss scale (Q,K)
  norm_rope_kernel<<<dim3(S_TOT), dim3(256), 0, stream>>>(Qf, Qbf, gq, fcos, fsin, SS_SCALE);
  norm_rope_kernel<<<dim3(S_TOT), dim3(256), 0, stream>>>(Kf, Kbf, gk, fcos, fsin, SS_SCALE);

  // transposed copies for K-contiguous access in aL/Y/aR GEMMs
  pack_kt_kernel<<<dim3((unsigned)((KTN + 255) / 256)), dim3(256), 0, stream>>>(Kf, Kt);
  pack_kt_kernel<<<dim3((unsigned)((KTN + 255) / 256)), dim3(256), 0, stream>>>(Vf, Vt);
  pack_qt_kernel<<<dim3((unsigned)((QTN + 255) / 256)), dim3(256), 0, stream>>>(Qf, Qt);

  // 4) monarch attention
  sum_frames_kernel<<<dim3((unsigned)((AR + 255) / 256)), dim3(256), 0, stream>>>(Qf, aR, aRbf);
  fill_kernel<<<dim3((HKJ + 255) / 256), dim3(256), 0, stream>>>(cRb, (float)FF, HKJ);

  GemmParams gbR = mk();   // bR: batch (h,k); C[j,(f,l)] = sum_d aR[k,j,h,:] K[f,k,l,h,:]
  gbR.M = GWD; gbR.N = FLN; gbR.Kd = HD; gbR.b2 = GHT; gbR.batches = NH * GHT;
  gbR.A = aRbf; gbR.sAb1 = sQh; gbR.sAb2 = sQi; gbR.sAm = sQj; gbR.sAk = 1; gbR.aVec = 1;
  gbR.B = Kbf;  gbR.sBb1 = sQh; gbR.sBb2 = sQi; gbR.sBk2 = 1; gbR.bVec = 1;
  gbR.divN = GWD; gbR.sBn1 = sQa; gbR.sBn2 = sQj;
  gbR.C = Rw; gbR.sCb1 = sRhF; gbR.sCb2 = sRkF; gbR.sCm2 = FLN; gbR.sCn = 1;

  GemmParams gaL = mk();   // aL: batch (h,k); C[j,d] = sum_{fl} R[.,fl] Kt[h,k,d,fl]
  gaL.M = GWD; gaL.N = HD; gaL.Kd = FLN; gaL.b2 = GHT; gaL.batches = NH * GHT;
  gaL.A = Rbf; gaL.sAb1 = sRhP; gaL.sAb2 = sRkP; gaL.sAm = FLP; gaL.sAk = 1; gaL.aVec = 1;
  gaL.B = Kt; gaL.sBb1 = (long)GHT * HD * FLP; gaL.sBb2 = (long)HD * FLP;
  gaL.sBk2 = 1; gaL.sBn2 = FLP; gaL.bVec = 1;
  gaL.C = aL; gaL.Cbf = aLbf;
  gaL.sCb1 = sQh; gaL.sCb2 = sQj; gaL.sCm2 = (long)GHT * NH * HD; gaL.sCn = 1;

  GemmParams gbL = mk();   // bL: batch (h,j); C[k,(a,i)] = sum_d aL[j,k,h,:] Q[a,i,j,h,:]
  gbL.M = GHT; gbL.N = AIN; gbL.Kd = HD; gbL.b2 = GWD; gbL.batches = NH * GWD;
  gbL.A = aLbf; gbL.sAb1 = sQh; gbL.sAb2 = (long)GHT * NH * HD; gbL.sAm = sQj; gbL.sAk = 1; gbL.aVec = 1;
  gbL.B = Qbf; gbL.sBb1 = sQh; gbL.sBb2 = sQj; gbL.sBk2 = 1; gbL.bVec = 1;
  gbL.divN = GHT; gbL.sBn1 = sQa; gbL.sBn2 = sQi;
  gbL.C = bLw; gbL.sCb1 = sLhF; gbL.sCb2 = sLjF; gbL.sCm2 = AIN; gbL.sCn = 1;

  GemmParams gaR = mk();   // aR: batch (h,j); C[k,d] = sum_{ai} L[.,ai] Qt[h,j,d,ai]
  gaR.M = GHT; gaR.N = HD; gaR.Kd = AIN; gaR.b2 = GWD; gaR.batches = NH * GWD;
  gaR.A = Lbf; gaR.sAb1 = sLhP; gaR.sAb2 = sLjP; gaR.sAm = AIP; gaR.sAk = 1; gaR.aVec = 1;
  gaR.B = Qt; gaR.sBb1 = (long)GWD * HD * AIP; gaR.sBb2 = (long)HD * AIP;
  gaR.sBk2 = 1; gaR.sBn2 = AIP; gaR.bVec = 1;
  gaR.C = aR; gaR.Cbf = aRbf;
  gaR.sCb1 = sQh; gaR.sCb2 = sQj; gaR.sCm2 = sQi; gaR.sCn = 1;

  GemmParams gY = gaL;     // Y: like aL but with Vt, C layout (k,j,h,d)
  gY.B = Vt; gY.C = Yw; gY.Cbf = Ybf;
  gY.sCb1 = sQh; gY.sCb2 = sQi; gY.sCm2 = sQj; gY.sCn = 1;

  GemmParams gO = mk();    // out: batch (h,j); C[(a,i),d] = sum_k L[h,j,k,(a,i)] Y[k,j,h,:]
  gO.M = AIN; gO.N = HD; gO.Kd = GHT; gO.b2 = GWD; gO.batches = NH * GWD;
  gO.A = Lbf; gO.sAb1 = sLhP; gO.sAb2 = sLjP; gO.sAm = 1; gO.sAk = AIP; gO.aVec = 0;
  gO.B = Ybf; gO.sBb1 = sQh; gO.sBb2 = sQj; gO.sBk2 = sQi; gO.sBn2 = 1; gO.bVec = 0;
  gO.C = attn; gO.Cbf = attnbf;
  gO.divM = GHT; gO.sCm1 = sQa; gO.sCm2 = sQi;
  gO.sCb1 = sQh; gO.sCb2 = sQj; gO.sCn = 1;

  // ---- iteration 1 (num_iters-1 = 1 refinement pass) ----
  run(gbR);
  softmax_R_kernel<<<dim3(HKJ), dim3(256), 0, stream>>>(Rw, Rbf, cRb, cLb);
  run(gaL);
  run(gbL);
  softmax_L_kernel<<<dim3(NH * GWD), dim3(256), 0, stream>>>(bLw, Lbf, cLb);
  rowsum_kernel<<<dim3(HKJ), dim3(256), 0, stream>>>(bLw, cRb);
  run(gaR);

  // ---- final right_half + output ----
  run(gbR);
  softmax_R_kernel<<<dim3(HKJ), dim3(256), 0, stream>>>(Rw, Rbf, cRb, cLb);
  run(gaL);
  run(gY);
  run(gbL);
  softmax_L_kernel<<<dim3(NH * GWD), dim3(256), 0, stream>>>(bLw, Lbf, cLb);
  run(gO);

  // 5) output projection: out = attn @ wo.T + bo
  {
    GemmParams p = mk();
    p.M = S_TOT; p.N = DIMC; p.Kd = DIMC;
    p.A = attnbf; p.sAm = DIMC; p.sAk = 1; p.aVec = 1;
    p.B = woBf; p.sBk2 = 1; p.sBn2 = DIMC; p.bVec = 1;
    p.C = out; p.sCm2 = DIMC; p.sCn = 1; p.bias = bo;
    run(p);
  }
  (void)in_sizes; (void)n_in; (void)out_size; (void)ws_size;
}
